// BalanceLabelAugmentation_5265629905347
// MI455X (gfx1250) — compile-verified, matrix operand order checked by reading, not run on hardware
//
#include <hip/hip_runtime.h>
#include <hip/hip_bf16.h>
#include <math.h>

// ---------------- problem constants (from reference) ----------------
#define NROWS 65536            // N
#define AFEAT 1024             // A
#define CCLS  1000             // C
#define SAUG  2                // S
#define MAUG  (SAUG * CCLS)    // 2000 augmented rows
#define MTOT  (NROWS + MAUG)   // 67536 total GEMM rows
#define CPAD  1008             // columns padded to 63 * 16
#define NTILE 63               // number of 16-column tiles
#define KCHUNK 512             // K per TDM stage (2 stages per n-tile)
#define NSTAGE (NTILE * 2)     // 126 pipeline stages

typedef __attribute__((ext_vector_type(16))) __bf16 v16bf;
typedef __attribute__((ext_vector_type(8)))  float  v8f;
typedef __attribute__((ext_vector_type(4)))  unsigned v4u;
typedef __attribute__((ext_vector_type(8)))  int    v8i;
typedef __attribute__((ext_vector_type(4)))  int    v4i;

static __device__ __forceinline__ unsigned short f2bf(float f) {
    union { float f; unsigned u; } v; v.f = f;
    unsigned u = v.u + 0x7FFFu + ((v.u >> 16) & 1u);   // round-to-nearest-even
    return (unsigned short)(u >> 16);
}

// ---------------- zero accumulators + output ----------------
__global__ void k_zero(float* gsum, float* gsq, float* cnt, float* out) {
    int i = blockIdx.x * 256 + threadIdx.x;
    if (i < CCLS * AFEAT) { gsum[i] = 0.f; gsq[i] = 0.f; }
    if (i < CCLS) cnt[i] = 0.f;
    if (i == 0) out[0] = 0.f;
}

// ---------------- per-class counts ----------------
__global__ void k_count(const int* __restrict__ lab, float* cnt) {
    int i = blockIdx.x * 256 + threadIdx.x;
    if (i < NROWS) atomicAdd(&cnt[lab[i]], 1.0f);
}

// ---------------- per-class sum / sum-of-squares (LDS-tiled) ----------------
// grid = 64 column-chunks x 8 row-chunks; LDS table [1000][16] x2 (128 KB)
__global__ void __launch_bounds__(256) k_stats(const float* __restrict__ x,
                                               const int* __restrict__ lab,
                                               float* gsum, float* gsq) {
    __shared__ float ls[CCLS * 16];
    __shared__ float lq[CCLS * 16];
    for (int i = threadIdx.x; i < CCLS * 16; i += 256) { ls[i] = 0.f; lq[i] = 0.f; }
    __syncthreads();

    const int cb = (blockIdx.x & 63) * 16;          // column base
    const int r0 = (blockIdx.x >> 6) * (NROWS / 8); // row-chunk base
    const int col = threadIdx.x & 15;
    const int rl  = threadIdx.x >> 4;
    for (int r = r0 + rl; r < r0 + NROWS / 8; r += 16) {
        int l = lab[r];
        float v = x[(size_t)r * AFEAT + cb + col];
        atomicAdd(&ls[l * 16 + col], v);       // ds_add_f32
        atomicAdd(&lq[l * 16 + col], v * v);
    }
    __syncthreads();
    for (int i = threadIdx.x; i < CCLS * 16; i += 256) {
        int c = i >> 4, a = cb + (i & 15);
        atomicAdd(&gsum[c * AFEAT + a], ls[i]);    // global_atomic_add_f32
        atomicAdd(&gsq [c * AFEAT + a], lq[i]);
    }
}

// ---------------- finalize stats, emit augmented rows ----------------
__global__ void k_final(const float* __restrict__ gsum, const float* __restrict__ gsq,
                        const float* __restrict__ cnt,
                        const float* __restrict__ covP, const float* __restrict__ aveP,
                        const float* __restrict__ amtP,
                        const float* __restrict__ eps, float* aug) {
    int i = blockIdx.x * 256 + threadIdx.x;
    if (i >= CCLS * AFEAT) return;
    int c = i / AFEAT, a = i - c * AFEAT;
    float n  = cnt[c];
    float nc = (n == 0.f) ? 1.f : n;
    float ave = gsum[i] / nc;
    float var = gsq[i] / nc - ave * ave;           // == E[(x-ave)^2] for n>0; 0 for n==0
    if (var <= 0.f) var = 1e-10f;
    float w = n / (n + amtP[c] + 1e-10f);
    float d = aveP[i] - ave;
    float ncov = covP[i] * (1.f - w) + var * w + w * (1.f - w) * d * d;
    float nave = aveP[i] * (1.f - w) + ave * w;
    aug[(size_t)(0 * CCLS + c) * AFEAT + a] = nave + ncov * eps[(size_t)(0 * CCLS + c) * AFEAT + a];
    aug[(size_t)(1 * CCLS + c) * AFEAT + a] = nave + ncov * eps[(size_t)(1 * CCLS + c) * AFEAT + a];
}

// ---------------- fc_w -> bf16 (padded), fused bias ----------------
__global__ void k_fcw(const float* __restrict__ fw, const float* __restrict__ fb,
                      const float* __restrict__ ba,
                      unsigned short* fcw16, float* biasP) {
    int i = blockIdx.x * 256 + threadIdx.x;
    if (i < CPAD * AFEAT) {
        int c = i / AFEAT;
        fcw16[i] = (c < CCLS) ? f2bf(fw[(size_t)c * AFEAT + (i - c * AFEAT)])
                              : (unsigned short)0;
    }
    if (i < CPAD) biasP[i] = (i < CCLS) ? (fb[i] - ba[i]) : -1e30f;
}

// ---------------- fused bf16-WMMA GEMM + online log-softmax + NLL ----------------
// block = 256 threads = 8 wave32 waves; block owns 128 rows.
// LDS: As 128x1024 bf16 = 256 KB + 2 TDM-double-buffered B tiles 16x512 bf16
//      (2 x 16 KB) = 288 KB  (<= 320 KB/WGP).
// B tiles stream in via TENSOR_LOAD_TO_LDS (TDM), pipelined one stage ahead.
__global__ void __launch_bounds__(256, 1) k_gemm(
        const float* __restrict__ x, const float* __restrict__ aug,
        const int* __restrict__ lab,
        const unsigned short* __restrict__ fcw16, const float* __restrict__ biasP,
        float* out) {
    __shared__ unsigned short As[128 * AFEAT];
    __shared__ unsigned short Bs[2][16 * KCHUNK];

    const int tid = threadIdx.x;
    const int rowBase = blockIdx.x * 128;

    // ---- stage A tile (f32 global -> bf16 LDS); zero-pad rows past MTOT ----
    for (int i = tid; i < (128 * AFEAT) / 4; i += 256) {
        int r  = i >> 8;              // 256 float4 per row
        int c4 = (i & 255) << 2;
        int rg = rowBase + r;
        float4 v = make_float4(0.f, 0.f, 0.f, 0.f);
        if (rg < NROWS)     v = *(const float4*)(x   + (size_t)rg * AFEAT + c4);
        else if (rg < MTOT) v = *(const float4*)(aug + (size_t)(rg - NROWS) * AFEAT + c4);
        unsigned short* p = &As[r * AFEAT + c4];
        p[0] = f2bf(v.x); p[1] = f2bf(v.y); p[2] = f2bf(v.z); p[3] = f2bf(v.w);
    }

    const int lane = tid & 31;
    const int wv   = tid >> 5;           // wave id 0..7 -> rows 16*wv .. 16*wv+15
    const int hlf  = lane >> 4;          // 0: rows M=0..7 ; 1: rows M=8..15 (C/D layout)
    const int rbase = rowBase + 16 * wv + (hlf << 3);

    // labels of the 8 rows tracked by this lane's half
    int lab8[8];
    for (int v = 0; v < 8; ++v) {
        int rg = rbase + v;
        lab8[v] = (rg < NROWS) ? lab[rg]
                : ((rg < MTOT) ? (rg - NROWS) % CCLS : -1);
    }
    float m8[8], s8[8], t8[8];
    for (int v = 0; v < 8; ++v) { m8[v] = -3.0e38f; s8[v] = 0.f; t8[v] = 0.f; }

    const int arow  = 16 * wv + (lane & 15);   // A frag: lane l holds row l%16
    const int bcolL = lane & 15;               // B frag: lane l holds col l%16
    const int kh    = hlf << 3;                // K-half offset (0 or 8) per ISA layout

    // ---- TDM descriptor pieces (uniform) ----
    const unsigned long long gbase = (unsigned long long)(size_t)fcw16;
    const unsigned ldsOff[2] = { (unsigned)(size_t)&Bs[0][0],
                                 (unsigned)(size_t)&Bs[1][0] };
    // issue one 2D TDM stage: 16 rows x 512 bf16, row stride 1024 bf16
    auto tdm_issue = [&](int s) {
        int nt = s >> 1, h = s & 1;
        unsigned long long ga = gbase
            + (((size_t)nt * 16) * AFEAT + (size_t)h * KCHUNK) * 2ull;
        // D# group0: count=1 | lds_addr | global_addr | type=2 (bits 127:126)
        v4u g0 = { 1u, ldsOff[s & 1], (unsigned)ga,
                   (unsigned)(ga >> 32) | 0x80000000u };
        // D# group1: data_size=2B; tensor_dim0=1024; tensor_dim1=16;
        //            tile_dim0=512; tile_dim1=16; dim0_stride=1024
        v8i g1 = { 0x00010000,              // workgroup_mask=0, data_size=1 (2B)
                   0x04000000,              // tensor_dim0[15:0]=1024 @ bits[31:16]
                   0x00100000,              // tensor_dim1[15:0]=16  @ bits[31:16]
                   0x02000000,              // tile_dim0=512 @ bits[31:16]
                   16,                      // tile_dim1=16
                   1024,                    // tensor_dim0_stride[31:0]
                   0, 0 };
        v4i g2 = { 0, 0, 0, 0 };            // 2-D tile: groups 2/3 unused
        v8i g4 = { 0, 0, 0, 0, 0, 0, 0, 0 };
        __builtin_amdgcn_tensor_load_to_lds(g0, g1, g2, g2, g4, 0);
    };

    __syncthreads();                         // A tile ready

    if (wv == 0) tdm_issue(0);               // prime the pipeline

    const v8f vzero = {};
    v8f acc = vzero;
    for (int s = 0; s < NSTAGE; ++s) {
        if (wv == 0) {
            if (s + 1 < NSTAGE) {
                tdm_issue(s + 1);                            // prefetch next stage
                __builtin_amdgcn_s_wait_tensorcnt(1);        // current stage done
            } else {
                __builtin_amdgcn_s_wait_tensorcnt(0);
            }
        }
        __syncthreads();                     // all waves: Bs[s&1] is ready

        const unsigned short* B = &Bs[s & 1][0];
        const int kg = (s & 1) * KCHUNK;     // K offset of this stage within A
        for (int k0 = 0; k0 < KCHUNK; k0 += 32) {
            // fragment = two contiguous 16B LDS loads: K base..+7 and base+16..+23
            union { uint4 u[2]; v16bf h; } fa, fbm;
            const uint4* ap = (const uint4*)&As[arow * AFEAT + kg + k0 + kh];
            fa.u[0] = ap[0]; fa.u[1] = ap[2];
            const uint4* bp = (const uint4*)&B[bcolL * KCHUNK + k0 + kh];
            fbm.u[0] = bp[0]; fbm.u[1] = bp[2];
            acc = __builtin_amdgcn_wmma_f32_16x16x32_bf16(
                      false, fa.h, false, fbm.h, (short)0, acc, false, false);
        }

        if (s & 1) {
            // n-tile complete: bias + online logsumexp + target gather
            const int n0 = (s >> 1) * 16;
            float bias = biasP[n0 + bcolL];
            for (int v = 0; v < 8; ++v) {
                float a = acc[v] + bias;
                float tm = a;
                for (int off = 1; off < 16; off <<= 1)
                    tm = fmaxf(tm, __shfl_xor(tm, off, 32));
                float nm = fmaxf(m8[v], tm);
                float e = __expf(a - nm);
                float ts = (lab8[v] == n0 + bcolL) ? a : 0.f;
                for (int off = 1; off < 16; off <<= 1) {
                    e  += __shfl_xor(e,  off, 32);
                    ts += __shfl_xor(ts, off, 32);
                }
                s8[v] = s8[v] * __expf(m8[v] - nm) + e;
                m8[v] = nm;
                t8[v] += ts;
            }
            acc = vzero;
        }
        __syncthreads();                     // done reading Bs[s&1] before reuse
    }

    // one lane per 16-lane group commits its 8 rows
    if ((lane & 15) == 0) {
        float part = 0.f;
        for (int v = 0; v < 8; ++v) {
            int rg = rbase + v;
            if (rg < MTOT) part += (m8[v] + __logf(s8[v])) - t8[v];
        }
        atomicAdd(out, part * (1.0f / (float)MTOT));
    }
}

// ---------------- driver ----------------
extern "C" void kernel_launch(void* const* d_in, const int* in_sizes, int n_in,
                              void* d_out, int out_size, void* d_ws, size_t ws_size,
                              hipStream_t stream) {
    (void)in_sizes; (void)n_in; (void)out_size; (void)ws_size;
    const float* x    = (const float*)d_in[0];
    const float* eps  = (const float*)d_in[1];
    const float* fcW  = (const float*)d_in[2];
    const float* fcB  = (const float*)d_in[3];
    const float* baC  = (const float*)d_in[4];
    const float* covP = (const float*)d_in[5];
    const float* aveP = (const float*)d_in[6];
    const float* amtP = (const float*)d_in[7];
    const int*   tgt  = (const int*)d_in[8];
    float* out = (float*)d_out;

    // workspace layout (floats): gsum | gsq | cnt | aug | biasP | fcw16(bf16)
    float* wsf  = (float*)d_ws;
    float* gsum = wsf;
    float* gsq  = gsum + (size_t)CCLS * AFEAT;
    float* cnt  = gsq  + (size_t)CCLS * AFEAT;
    float* aug  = cnt  + CCLS;
    float* biasP = aug + (size_t)MAUG * AFEAT;
    unsigned short* fcw16 = (unsigned short*)(biasP + CPAD);

    k_zero <<<4000, 256, 0, stream>>>(gsum, gsq, cnt, out);
    k_count<<<NROWS / 256, 256, 0, stream>>>(tgt, cnt);
    k_stats<<<512, 256, 0, stream>>>(x, tgt, gsum, gsq);
    k_final<<<4000, 256, 0, stream>>>(gsum, gsq, cnt, covP, aveP, amtP, eps, aug);
    k_fcw  <<<(CPAD * AFEAT) / 256, 256, 0, stream>>>(fcW, fcB, baC, fcw16, biasP);
    k_gemm <<<(MTOT + 127) / 128, 256, 0, stream>>>(x, aug, tgt, fcw16, biasP, out);
}
